// ConDivNet_27066883899773
// MI455X (gfx1250) — matrix operand order repaired and checked by reading, not compiled, not run on hardware
//
#include <hip/hip_runtime.h>

typedef __attribute__((ext_vector_type(16))) _Float16 v16h;
typedef __attribute__((ext_vector_type(8)))  _Float16 v8h;
typedef __attribute__((ext_vector_type(8)))  float    v8f;
typedef int v4i_vs __attribute__((vector_size(16)));   // matches async builtin param type

#define LDSTRIDE 1032   // 1024 + 8 halves padding; row stride = 2064 B (16B aligned)
#define T_STEPS  200
#define BATCH    128
#define NTOT     3072   // 256 + 1024 + 512 + 1024 + 256

// ---- workspace layout (bytes) ----
#define WGT_HALVES   1572864
#define WS_PAR_OFF   3145728
#define WS_ST_OFF    (WS_PAR_OFF + 4*NTOT*4)
#define ST_FLOATS    (BATCH*NTOT)          // 393216 floats per state array
#define PREP_N       (3*ST_FLOATS + BATCH) // state floats to zero each launch

// output offsets (floats)
#define OFF_MR   0
#define OFF_SIN  25600
#define OFF_SE1  6579200
#define OFF_SB   32793600
#define OFF_SE2  45900800
#define OFF_SOUT 72115200

// weight bases (halves) and neuron bases
#define WB_E1 0
#define WB_B  262144
#define WB_E2 786432
#define WB_O  1310720
#define NB_IN 0
#define NB_E1 256
#define NB_B  1280
#define NB_E2 1792
#define NB_O  2816

// ---- CDNA5 async global->LDS path ----
#if defined(__HIP_DEVICE_COMPILE__)
# if __has_builtin(__builtin_amdgcn_global_load_async_to_lds_b128) && \
     __has_builtin(__builtin_amdgcn_s_wait_asynccnt)
#  define HAVE_ASYNC 1
# endif
#endif
#ifndef HAVE_ASYNC
# define HAVE_ASYNC 0
#endif

#if HAVE_ASYNC
# define ASYNC_CP16(gsrc, ldst)                                             \
    __builtin_amdgcn_global_load_async_to_lds_b128(                         \
        (__attribute__((address_space(1))) v4i_vs*)(gsrc),                  \
        (__attribute__((address_space(3))) v4i_vs*)(ldst), 0, 0)
# define WAIT_ASYNC(n) __builtin_amdgcn_s_wait_asynccnt(n)
#endif

__device__ __forceinline__ float clampab(float v) {
    return fminf(fmaxf(v, 1e-6f), 1.0f);
}

// ---------------- prep: weights->f16, neuron constants, zero state ----------------
__global__ void snn_prep(const float* __restrict__ wE1, const float* __restrict__ wB,
                         const float* __restrict__ wE2, const float* __restrict__ wO,
                         const float* __restrict__ aI, const float* __restrict__ bI, const float* __restrict__ tI,
                         const float* __restrict__ a1, const float* __restrict__ b1, const float* __restrict__ t1,
                         const float* __restrict__ aB, const float* __restrict__ bB, const float* __restrict__ tB,
                         const float* __restrict__ a2, const float* __restrict__ b2, const float* __restrict__ t2,
                         const float* __restrict__ aO, const float* __restrict__ bO, const float* __restrict__ tO,
                         _Float16* __restrict__ W16, float* __restrict__ PAR, float* __restrict__ ST)
{
    int g = blockIdx.x * blockDim.x + threadIdx.x;
    if (g < WGT_HALVES) {
        float w;
        if      (g < WB_B)  w = wE1[g];
        else if (g < WB_E2) w = wB [g - WB_B];
        else if (g < WB_O)  w = wE2[g - WB_E2];
        else                w = wO [g - WB_O];
        W16[g] = (_Float16)w;
    }
    if (g < PREP_N) ST[g] = 0.0f;   // p, q, m, mr zeroed every launch (determinism)
    if (g < NTOT) {
        float a, b, th;
        if      (g < NB_E1) { int i = g;          a = aI[i]; b = bI[i]; th = tI[i]; }
        else if (g < NB_B)  { int i = g - NB_E1;  a = a1[i]; b = b1[i]; th = t1[i]; }
        else if (g < NB_E2) { int i = g - NB_B;   a = aB[i]; b = bB[i]; th = tB[i]; }
        else if (g < NB_O)  { int i = g - NB_E2;  a = a2[i]; b = b2[i]; th = t2[i]; }
        else                { int i = g - NB_O;   a = aO[i]; b = bO[i]; th = tO[i]; }
        float aC = clampab(a), bC = clampab(b);
        float la = __logf(aC), lb = __logf(bC);
        PAR[g]          = aC;
        PAR[g + NTOT]   = bC;
        PAR[g + 2*NTOT] = lb / (lb - la);   // tau
        PAR[g + 3*NTOT] = th;
    }
}

// -------- one dense layer: spikes(LDS f16) @ W.T + bias -> alpha neuron -> spikes --------
template <int K, int N>
__device__ __forceinline__ void layer_gemm(
    const _Float16* __restrict__ W,       // [N][K] f16, L2-resident
    const float* __restrict__ bias,       // [N]
    int neuronBase,
    const _Float16* __restrict__ src,     // LDS spike tile [16][LDSTRIDE]
    _Float16* __restrict__ dst,           // LDS spike tile [16][LDSTRIDE]
    float* __restrict__ outBase,          // d_out + layer_off + (t*128 + r0)*N
    const float* __restrict__ Pa, const float* __restrict__ Pb,
    const float* __restrict__ Pt, const float* __restrict__ Pth,
    float* __restrict__ Sp, float* __restrict__ Sq, float* __restrict__ Sm,
    _Float16* __restrict__ stage,         // LDS staging: 8 waves x 2 bufs x 512 halves
    int r0)
{
    const int lane = threadIdx.x & 31;
    const int warp = threadIdx.x >> 5;
    const int hi   = lane >> 4;     // half-wave select
    const int lm   = lane & 15;
    constexpr int NK = K / 32;      // 8 / 16 / 32 -- always even, >= 4
    static_assert(NK >= 4 && (NK & 1) == 0, "pipeline assumes even NK >= 4");

    for (int n = warp; n < (N >> 4); n += 8) {        // uniform per wave -> EXEC stays all-ones
        const int ncol   = (n << 4) + lm;
        const int neuron = neuronBase + ncol;
        const float aC  = Pa[neuron], bC = Pb[neuron];
        const float tau = Pt[neuron], thr = Pth[neuron];
        const float bia = bias[ncol];

        const _Float16* wrow = W + (size_t)ncol * K + (hi << 4);        // B frag: lane=col, K=16*hi+0..15
        const _Float16* arow = src + (size_t)lm * LDSTRIDE + (hi << 4); // A frag: lane=row, K=16*hi+0..15

        v8f acc0 = {}, acc1 = {};     // two chains: consecutive WMMAs independent (hide D->A/B hazard)
#if HAVE_ASYNC
        // 2-deep, branch-free software pipeline of B-fragments through LDS (ASYNCcnt DMA).
        _Float16* stg = stage + (size_t)warp * 1024;          // 2 bufs x 512 halves
        _Float16* my0 = stg + lane * 16;
        _Float16* my1 = stg + 512 + lane * 16;
        ASYNC_CP16(wrow,      my0);
        ASYNC_CP16(wrow + 8,  my0 + 8);
        ASYNC_CP16(wrow + 32, my1);
        ASYNC_CP16(wrow + 40, my1 + 8);
        #pragma unroll 2
        for (int kb = 0; kb < NK - 2; ++kb) {         // steady state: no branches
            const int k0 = kb * 32;
            union { v16h v; v8h h[2]; } A, B;
            A.h[0] = *(const v8h*)(arow + k0);
            A.h[1] = *(const v8h*)(arow + k0 + 8);
            WAIT_ASYNC(2);
            _Float16* cb = (kb & 1) ? my1 : my0;
            B.h[0] = *(const v8h*)(cb);
            B.h[1] = *(const v8h*)(cb + 8);
            if (kb & 1)
                acc1 = __builtin_amdgcn_wmma_f32_16x16x32_f16(false, A.v, false, B.v, (short)0, acc1, false, false);
            else
                acc0 = __builtin_amdgcn_wmma_f32_16x16x32_f16(false, A.v, false, B.v, (short)0, acc0, false, false);
            const _Float16* gs = wrow + (kb + 2) * 32;    // refill the buffer just consumed
            ASYNC_CP16(gs,     cb);
            ASYNC_CP16(gs + 8, cb + 8);
        }
        {   // peeled: kb = NK-2 (even -> buf0/acc0), no refill
            constexpr int k0 = (NK - 2) * 32;
            union { v16h v; v8h h[2]; } A, B;
            A.h[0] = *(const v8h*)(arow + k0);
            A.h[1] = *(const v8h*)(arow + k0 + 8);
            WAIT_ASYNC(2);
            B.h[0] = *(const v8h*)(my0);
            B.h[1] = *(const v8h*)(my0 + 8);
            acc0 = __builtin_amdgcn_wmma_f32_16x16x32_f16(false, A.v, false, B.v, (short)0, acc0, false, false);
        }
        {   // peeled: kb = NK-1 (odd -> buf1/acc1), drain
            constexpr int k0 = (NK - 1) * 32;
            union { v16h v; v8h h[2]; } A, B;
            A.h[0] = *(const v8h*)(arow + k0);
            A.h[1] = *(const v8h*)(arow + k0 + 8);
            WAIT_ASYNC(0);
            B.h[0] = *(const v8h*)(my1);
            B.h[1] = *(const v8h*)(my1 + 8);
            acc1 = __builtin_amdgcn_wmma_f32_16x16x32_f16(false, A.v, false, B.v, (short)0, acc1, false, false);
        }
#else
        #pragma unroll 4
        for (int k0 = 0; k0 < K; k0 += 32) {
            union { v16h v; v8h h[2]; } A, B;
            A.h[0] = *(const v8h*)(arow + k0);
            A.h[1] = *(const v8h*)(arow + k0 + 8);
            B.h[0] = *(const v8h*)(wrow + k0);
            B.h[1] = *(const v8h*)(wrow + k0 + 8);
            __builtin_prefetch((const void*)(wrow + k0 + 256), 0, 0);
            if ((k0 >> 5) & 1)
                acc1 = __builtin_amdgcn_wmma_f32_16x16x32_f16(false, A.v, false, B.v, (short)0, acc1, false, false);
            else
                acc0 = __builtin_amdgcn_wmma_f32_16x16x32_f16(false, A.v, false, B.v, (short)0, acc0, false, false);
        }
#endif

        // D layout: vgpr e -> row m = e + 8*hi, col = ncol.  State rows contiguous per lane.
        const size_t sbase = (size_t)neuron * BATCH + r0 + (hi << 3);
        v8f P = *(const v8f*)(Sp + sbase);
        v8f Q = *(const v8f*)(Sq + sbase);
        v8f M = *(const v8f*)(Sm + sbase);
        #pragma unroll
        for (int e = 0; e < 8; ++e) {
            const int mrow = e + (hi << 3);
            const float cur = acc0[e] + acc1[e] + bia;
            float p = P[e], q = Q[e], mm = M[e];
            const float reset = (mm - thr > 0.0f) ? thr : 0.0f;   // from *previous* mem
            p  = aC * p + cur;
            q  = bC * q - cur;
            mm = tau * (p + q) - reset;
            const float spk = (mm - thr > 0.0f) ? 1.0f : 0.0f;
            P[e] = p; Q[e] = q; M[e] = mm;
            dst[(size_t)mrow * LDSTRIDE + ncol] = (_Float16)spk;
            __builtin_nontemporal_store(spk, outBase + (size_t)mrow * N + ncol);
        }
        *(v8f*)(Sp + sbase) = P;
        *(v8f*)(Sq + sbase) = Q;
        *(v8f*)(Sm + sbase) = M;
    }
}

// -------- persistent per-batch-slice kernel: whole T loop, no inter-WG sync --------
__global__ void __launch_bounds__(256)
snn_main(const float* __restrict__ x,
         const float* __restrict__ bE1, const float* __restrict__ bB,
         const float* __restrict__ bE2, const float* __restrict__ bO,
         const float* __restrict__ Wro, const float* __restrict__ bro,
         const float* __restrict__ betaro,
         const _Float16* __restrict__ W16,
         const float* __restrict__ PAR,
         float* __restrict__ ST,
         float* __restrict__ out)
{
    __shared__ __align__(16) _Float16 sA[16 * LDSTRIDE];
    __shared__ __align__(16) _Float16 sB[16 * LDSTRIDE];
    __shared__ __align__(16) _Float16 sStage[8 * 2 * 512];   // async B-fragment staging

    const int r0 = blockIdx.x * 16;           // this WG's batch-row slice
    const float* Pa  = PAR;
    const float* Pb  = PAR + NTOT;
    const float* Pt  = PAR + 2*NTOT;
    const float* Pth = PAR + 3*NTOT;
    float* Sp = ST;
    float* Sq = ST + ST_FLOATS;
    float* Sm = ST + 2*(size_t)ST_FLOATS;
    float* MR = ST + 3*(size_t)ST_FLOATS;
    const float bro0  = bro[0];
    const float betaR = fminf(fmaxf(betaro[0], 0.0f), 1.0f);

    for (int t = 0; t < T_STEPS; ++t) {
        const size_t tb = (size_t)t * BATCH + r0;

        // ---- input layer: cur = x[t,row] broadcast over 256 neurons ----
        for (int idx = threadIdx.x; idx < 16 * 256; idx += 256) {
            const int rl = idx & 15, ncol = idx >> 4;   // rl fastest -> coalesced state access
            const int rowg = r0 + rl;
            const float cur = x[(size_t)t * BATCH + rowg];
            const float aC = Pa[ncol], bC = Pb[ncol], tau = Pt[ncol], thr = Pth[ncol];
            const size_t sidx = (size_t)ncol * BATCH + rowg;
            float p = Sp[sidx], q = Sq[sidx], mm = Sm[sidx];
            const float reset = (mm - thr > 0.0f) ? thr : 0.0f;
            p  = aC * p + cur;
            q  = bC * q - cur;
            mm = tau * (p + q) - reset;
            const float spk = (mm - thr > 0.0f) ? 1.0f : 0.0f;
            Sp[sidx] = p; Sq[sidx] = q; Sm[sidx] = mm;
            sA[(size_t)rl * LDSTRIDE + ncol] = (_Float16)spk;
            __builtin_nontemporal_store(spk, out + OFF_SIN + tb * 256 + (size_t)rl * 256 + ncol);
        }
        __syncthreads();

        layer_gemm<256, 1024>(W16 + WB_E1, bE1, NB_E1, sA, sB,
                              out + OFF_SE1 + tb * 1024, Pa, Pb, Pt, Pth, Sp, Sq, Sm, sStage, r0);
        __syncthreads();
        layer_gemm<1024, 512>(W16 + WB_B, bB, NB_B, sB, sA,
                              out + OFF_SB + tb * 512, Pa, Pb, Pt, Pth, Sp, Sq, Sm, sStage, r0);
        __syncthreads();
        layer_gemm<512, 1024>(W16 + WB_E2, bE2, NB_E2, sA, sB,
                              out + OFF_SE2 + tb * 1024, Pa, Pb, Pt, Pth, Sp, Sq, Sm, sStage, r0);
        __syncthreads();
        layer_gemm<1024, 256>(W16 + WB_O, bO, NB_O, sB, sA,
                              out + OFF_SOUT + tb * 256, Pa, Pb, Pt, Pth, Sp, Sq, Sm, sStage, r0);
        __syncthreads();

        // ---- readout: mr = betaR*mr + spk_out . Wro + bro ----
        if (threadIdx.x < 16) {
            const int rl = threadIdx.x, rowg = r0 + rl;
            float acc = bro0;
            #pragma unroll 8
            for (int j = 0; j < 256; ++j)
                acc += (float)sA[(size_t)rl * LDSTRIDE + j] * Wro[j];
            const float mrv = betaR * MR[rowg] + acc;
            MR[rowg] = mrv;
            __builtin_nontemporal_store(mrv, out + OFF_MR + (size_t)t * BATCH + rowg);
        }
        __syncthreads();   // protect sA before next step's input stage rewrites it
    }
}

extern "C" void kernel_launch(void* const* d_in, const int* in_sizes, int n_in,
                              void* d_out, int out_size, void* d_ws, size_t ws_size,
                              hipStream_t stream) {
    const float* x   = (const float*)d_in[0];
    const float* wE1 = (const float*)d_in[1];  const float* bE1 = (const float*)d_in[2];
    const float* wB  = (const float*)d_in[3];  const float* bB  = (const float*)d_in[4];
    const float* wE2 = (const float*)d_in[5];  const float* bE2 = (const float*)d_in[6];
    const float* wO  = (const float*)d_in[7];  const float* bO  = (const float*)d_in[8];
    const float* Wro = (const float*)d_in[9];  const float* bro = (const float*)d_in[10];
    const float* aI = (const float*)d_in[11], *bI = (const float*)d_in[12], *tI = (const float*)d_in[13];
    const float* a1 = (const float*)d_in[14], *b1 = (const float*)d_in[15], *t1 = (const float*)d_in[16];
    const float* aB = (const float*)d_in[17], *bbB = (const float*)d_in[18], *tB = (const float*)d_in[19];
    const float* a2 = (const float*)d_in[20], *b2 = (const float*)d_in[21], *t2 = (const float*)d_in[22];
    const float* aO = (const float*)d_in[23], *bO2 = (const float*)d_in[24], *tO = (const float*)d_in[25];
    const float* betaro = (const float*)d_in[26];

    char* ws = (char*)d_ws;
    _Float16* W16 = (_Float16*)ws;
    float* PAR = (float*)(ws + WS_PAR_OFF);
    float* ST  = (float*)(ws + WS_ST_OFF);

    const int prepBlocks = (PREP_N + 255) / 256;
    snn_prep<<<prepBlocks, 256, 0, stream>>>(wE1, wB, wE2, wO,
                                             aI, bI, tI, a1, b1, t1,
                                             aB, bbB, tB, a2, b2, t2,
                                             aO, bO2, tO,
                                             W16, PAR, ST);

    snn_main<<<BATCH / 16, 256, 0, stream>>>(x, bE1, bB, bE2, bO, Wro, bro, betaro,
                                             W16, PAR, ST, (float*)d_out);
}
